// PhysicalStructuralModel_37486474560196
// MI455X (gfx1250) — matrix operand order repaired and checked by reading, not compiled, not run on hardware
//
#include <hip/hip_runtime.h>
#include <math.h>

typedef _Float16 v16h __attribute__((ext_vector_type(16)));
typedef _Float16 v8h  __attribute__((ext_vector_type(8)));
typedef _Float16 v4h  __attribute__((ext_vector_type(4)));
typedef float    v8f  __attribute__((ext_vector_type(8)));

#define NB 32
#define NTT 2048
#define NV 8
#define NH 5
#define ND 128
#define BT (NB*NTT)         // 65536 tokens
#define TPG 32              // tokens per wave-group (32 rows = 2 M-tiles)
#define GROUPS (BT/TPG)     // 2048
#define NW 8
#define THREADS (NW*32)
#define BLOCKS 256          // 256*8 waves = 2048 groups, 1 group/wave
#define HJ (NV*NH)          // 40
#define HK 64               // history K padded 40 -> 64 for WMMA

__device__ __forceinline__ v16h cat8(v8h lo, v8h hi) {
  return __builtin_shufflevector(lo, hi, 0,1,2,3,4,5,6,7,8,9,10,11,12,13,14,15);
}

// A fragment (16x32 f16) from row-major [16][stride] LDS tile.
__device__ __forceinline__ v16h load_a(const _Float16* t, int stride, int lane, int ks) {
  const _Float16* p = t + (lane & 15)*stride + ks*32 + (lane >> 4)*8;
  v8h lo = *(const v8h*)p;
  v8h hi = *(const v8h*)(p + 16);
  return cat8(lo, hi);
}

// B fragment (32x16 f16) where B[k][n] = W[n][k], W row-major in LDS.
__device__ __forceinline__ v16h load_b(const _Float16* w, int stride, int lane, int nt, int ks) {
  const _Float16* p = w + (nt*16 + (lane & 15))*stride + ks*32 + (lane >> 4)*8;
  v8h lo = *(const v8h*)p;
  v8h hi = *(const v8h*)(p + 16);
  return cat8(lo, hi);
}

// Dual-M software-pipelined GEMM: 32 rows (two 16-row M-tiles share every B
// fragment -> 2 WMMAs per B ds_load pair, halving LDS traffic per matrix op).
template<int KSN, int NTN>
__device__ __forceinline__ void gemm_dual(const _Float16* A, int astr,
                                          const _Float16* W, int bstr,
                                          v8f* accA, v8f* accB, int lane) {
  const _Float16* A1 = A + 16*astr;
  v16h a0 = load_a(A,  astr, lane, 0);
  v16h a1 = load_a(A1, astr, lane, 0);
  v16h b  = load_b(W,  bstr, lane, 0, 0);
  #pragma unroll
  for (int t = 0; t < KSN*NTN; ++t) {
    const int ks = t / NTN, nt = t % NTN;
    v16h bn = b, a0n = a0, a1n = a1;
    if (t + 1 < KSN*NTN)
      bn = load_b(W, bstr, lane, (t+1) % NTN, (t+1) / NTN);
    if (nt == NTN-1 && ks < KSN-1) {
      a0n = load_a(A,  astr, lane, ks+1);
      a1n = load_a(A1, astr, lane, ks+1);
    }
    accA[nt] = __builtin_amdgcn_wmma_f32_16x16x32_f16(
        false, a0, false, b, (short)0, accA[nt], false, false);
    accB[nt] = __builtin_amdgcn_wmma_f32_16x16x32_f16(
        false, a1, false, b, (short)0, accB[nt], false, false);
    b = bn; a0 = a0n; a1 = a1n;
  }
}

// Single-M pipelined GEMM (for the 16x16 score blocks).
template<int KSN, int NTN>
__device__ __forceinline__ void gemm_pipe(const _Float16* A, int astr,
                                          const _Float16* W, int bstr,
                                          v8f* acc, int lane) {
  v16h a_cur = load_a(A, astr, lane, 0);
  v16h b_cur = load_b(W, bstr, lane, 0, 0);
  #pragma unroll
  for (int t = 0; t < KSN*NTN; ++t) {
    const int ks = t / NTN, nt = t % NTN;
    v16h b_nxt = b_cur, a_nxt = a_cur;
    if (t + 1 < KSN*NTN)
      b_nxt = load_b(W, bstr, lane, (t+1) % NTN, (t+1) / NTN);
    if (nt == NTN-1 && ks < KSN-1)
      a_nxt = load_a(A, astr, lane, ks+1);
    acc[nt] = __builtin_amdgcn_wmma_f32_16x16x32_f16(
        false, a_cur, false, b_cur, (short)0, acc[nt], false, false);
    b_cur = b_nxt; a_cur = a_nxt;
  }
}

// Scatter C accumulators to row-major f16 LDS tile (+bias, optional relu).
// C layout: lanes 0-15: row=i, col=lane ; lanes 16-31: row=i+8, col=lane-16.
__device__ __forceinline__ void store_c(_Float16* t, const v8f* acc, int nacc, int lane,
                                        const float* bias, bool relu) {
  int cl = lane & 15, half = lane >> 4;
  for (int nt = 0; nt < nacc; ++nt) {
    int col = nt*16 + cl;
    float bv = bias ? bias[col] : 0.f;
    #pragma unroll
    for (int i = 0; i < 8; ++i) {
      int row = i + 8*half;
      float v = acc[nt][i] + bv;
      if (relu) v = v > 0.f ? v : 0.f;
      t[row*ND + col] = (_Float16)v;
    }
  }
}

__global__ __launch_bounds__(THREADS)
void psm_fused_kernel(const float* __restrict__ x, const float* __restrict__ history,
                      const float* __restrict__ ce_w1, const float* __restrict__ ce_b1,
                      const float* __restrict__ ce_w2, const float* __restrict__ ce_b2,
                      const float* __restrict__ p_w1, const float* __restrict__ p_b1,
                      const float* __restrict__ p_w2, const float* __restrict__ p_b2,
                      const float* __restrict__ wq, const float* __restrict__ wk,
                      float* __restrict__ out_pred, float* __restrict__ out_A) {
  // f16 weights resident for the whole block (128 KB)
  __shared__ __align__(16) _Float16 wgt[3][ND*ND];   // p_w2, wq, wk
  __shared__ __align__(16) _Float16 cw1[ND*HK];      // ce_w1 K-padded
  __shared__ __align__(16) _Float16 cw2[64*ND];      // ce_w2
  __shared__ float pw1s[ND], pb1s[ND], pb2s[ND], cb1s[ND], cb2s[64];
  // per-wave scratch; context phase overlays: hist lives in tA, h lives in tB
  __shared__ __align__(16) _Float16 tA[NW][TPG*ND];  // hist | feat1 | q   (8 KB/wave)
  __shared__ __align__(16) _Float16 tB[NW][TPG*ND];  // h | feat2 | k      (8 KB/wave)
  __shared__ _Float16 gbs[NW][TPG*64];               // gate (f16)         (4 KB/wave)
  __shared__ float xbs[NW][TPG*NV];                  // x                  (1 KB/wave)
  __shared__ float scs[NW][TPG*16];                  //                    (2 KB/wave)

  const int tid = threadIdx.x;
  const int w = tid >> 5, lane = tid & 31;
  const int cl = lane & 15, half = lane >> 4;

  // ---- stage + fp32->f16 convert all weights into LDS ----
  for (int i = tid; i < ND*ND; i += THREADS) {
    wgt[0][i] = (_Float16)p_w2[i];
    wgt[1][i] = (_Float16)wq[i];
    wgt[2][i] = (_Float16)wk[i];
  }
  for (int i = tid; i < ND*HK; i += THREADS) {
    int d = i >> 6, c = i & (HK-1);
    cw1[i] = (_Float16)(c < HJ ? ce_w1[d*HJ + c] : 0.f);
  }
  for (int i = tid; i < 64*ND; i += THREADS)
    cw2[i] = (_Float16)ce_w2[i];
  if (tid < ND) { pw1s[tid] = p_w1[tid]; pb1s[tid] = p_b1[tid];
                  pb2s[tid] = p_b2[tid]; cb1s[tid] = ce_b1[tid]; }
  if (tid < 64)  cb2s[tid] = ce_b2[tid];
  __syncthreads();

  v8f zero = {};
  v8f accA[8], accB[8];

  for (int group = blockIdx.x*NW + w; group < GROUPS; group += BLOCKS*NW) {
    __builtin_prefetch(&history[((size_t)(group + BLOCKS*NW)*TPG*HJ) % ((size_t)BT*HJ)], 0, 0);

    // ---- stage history (32x40 -> padded 32x64 f16, in tA) and x (32x8) ----
    for (int i = lane; i < TPG*HK; i += 32) {
      int r = i >> 6, c = i & (HK-1);
      tA[w][i] = (_Float16)(c < HJ ? history[((size_t)group*TPG + r)*HJ + c] : 0.f);
    }
    for (int i = lane; i < TPG*NV; i += 32)
      xbs[w][i] = x[(size_t)group*TPG*NV + i];
    asm volatile("s_wait_dscnt 0" ::: "memory");

    // ---- h = relu(hist @ ce_w1^T + b1) : 32 WMMA, h -> tB ----
    #pragma unroll
    for (int i = 0; i < 8; ++i) { accA[i] = zero; accB[i] = zero; }
    gemm_dual<2,8>(tA[w], HK, cw1, HK, accA, accB, lane);
    store_c(tB[w],        accA, 8, lane, cb1s, true);
    store_c(tB[w]+16*ND,  accB, 8, lane, cb1s, true);
    asm volatile("s_wait_dscnt 0" ::: "memory");

    // ---- gate = sigmoid(h @ ce_w2^T + b2) : 32 WMMA ----
    #pragma unroll
    for (int i = 0; i < 4; ++i) { accA[i] = zero; accB[i] = zero; }
    gemm_dual<4,4>(tB[w], ND, cw2, ND, accA, accB, lane);
    #pragma unroll
    for (int nt = 0; nt < 4; ++nt) {
      int col = nt*16 + cl;              // col = i*8+j gate index
      #pragma unroll
      for (int i = 0; i < 8; ++i) {
        int row = i + 8*half;            // token-in-group (sub-tile 0 / 1)
        gbs[w][row*64 + col]      = (_Float16)(1.f / (1.f + __expf(-(accA[nt][i] + cb2s[col]))));
        gbs[w][(row+16)*64 + col] = (_Float16)(1.f / (1.f + __expf(-(accB[nt][i] + cb2s[col]))));
      }
    }
    asm volatile("s_wait_dscnt 0" ::: "memory");

    // ---- 8 quad-tiles: tokens (4*st .. 4*st+3), 32 rows each ----
    for (int st = 0; st < 8; ++st) {
      // feat1 = relu(x * p_w1 + b1) -> tA
      #pragma unroll
      for (int r = 0; r < TPG; ++r) {
        float xv = xbs[w][st*TPG + r];   // (st*4 + r>>3)*8 + (r&7)
        int d0 = lane*4;
        v4h pk;
        #pragma unroll
        for (int j = 0; j < 4; ++j) {
          int d = d0 + j;
          float f = xv*pw1s[d] + pb1s[d];
          pk[j] = (_Float16)(f > 0.f ? f : 0.f);
        }
        *(v4h*)&tA[w][r*ND + d0] = pk;
      }
      asm volatile("s_wait_dscnt 0" ::: "memory");

      // GEMM1: feat2 = feat1 @ p_w2^T + b2 -> tB   (64 WMMA)
      #pragma unroll
      for (int i = 0; i < 8; ++i) { accA[i] = zero; accB[i] = zero; }
      gemm_dual<4,8>(tA[w], ND, wgt[0], ND, accA, accB, lane);
      store_c(tB[w],       accA, 8, lane, pb2s, false);
      store_c(tB[w]+16*ND, accB, 8, lane, pb2s, false);
      asm volatile("s_wait_dscnt 0" ::: "memory");

      // GEMM2: q = feat2 @ wq^T -> tA              (64 WMMA)
      #pragma unroll
      for (int i = 0; i < 8; ++i) { accA[i] = zero; accB[i] = zero; }
      gemm_dual<4,8>(tB[w], ND, wgt[1], ND, accA, accB, lane);
      store_c(tA[w],       accA, 8, lane, nullptr, false);
      store_c(tA[w]+16*ND, accB, 8, lane, nullptr, false);
      asm volatile("s_wait_dscnt 0" ::: "memory");

      // GEMM3: k = feat2 @ wk^T -> tB              (64 WMMA)
      #pragma unroll
      for (int i = 0; i < 8; ++i) { accA[i] = zero; accB[i] = zero; }
      gemm_dual<4,8>(tB[w], ND, wgt[2], ND, accA, accB, lane);
      store_c(tB[w],       accA, 8, lane, nullptr, false);
      store_c(tB[w]+16*ND, accB, 8, lane, nullptr, false);
      asm volatile("s_wait_dscnt 0" ::: "memory");

      // scores + epilogue per 16-row sub-tile (2 tokens each)
      #pragma unroll
      for (int s2 = 0; s2 < 2; ++s2) {
        v8f sacc[1];
        sacc[0] = zero;
        gemm_pipe<4,1>(tA[w] + s2*16*ND, ND, tB[w] + s2*16*ND, ND, sacc, lane);

        int tc = cl >> 3;
        #pragma unroll
        for (int i = 0; i < 8; ++i) {
          int row = i + 8*half;
          int tr = row >> 3;
          float a = tanhf(sacc[0][i] * 0.08838834764831845f);  // 1/sqrt(128)
          float part = 0.f;
          if (tr == tc) {
            int lt = st*4 + s2*2 + tr;   // token in group
            float g  = (float)gbs[w][lt*64 + (row & 7)*8 + (cl & 7)];
            float av = a * g;
            out_A[(((size_t)group*TPG + lt)*NV + (row & 7))*NV + (cl & 7)] = av;
            part = av * xbs[w][st*TPG + s2*16 + cl];
          }
          scs[w][(s2*16 + row)*16 + cl] = part;
        }
        asm volatile("s_wait_dscnt 0" ::: "memory");

        if (lane < 16) {
          int row = lane, tk = row >> 3;
          float p = 0.f;
          #pragma unroll
          for (int c = 0; c < 8; ++c) p += scs[w][(s2*16 + row)*16 + tk*8 + c];
          out_pred[((size_t)group*TPG + st*4 + s2*2 + tk)*NV + (row & 7)] = p;
        }
        asm volatile("s_wait_dscnt 0" ::: "memory");
      }
    }
  }
}

extern "C" void kernel_launch(void* const* d_in, const int* in_sizes, int n_in,
                              void* d_out, int out_size, void* d_ws, size_t ws_size,
                              hipStream_t stream) {
  const float* x     = (const float*)d_in[0];
  const float* hist  = (const float*)d_in[1];
  const float* ce_w1 = (const float*)d_in[2];
  const float* ce_b1 = (const float*)d_in[3];
  const float* ce_w2 = (const float*)d_in[4];
  const float* ce_b2 = (const float*)d_in[5];
  const float* p_w1  = (const float*)d_in[6];
  const float* p_b1  = (const float*)d_in[7];
  const float* p_w2  = (const float*)d_in[8];
  const float* p_b2  = (const float*)d_in[9];
  const float* wqp   = (const float*)d_in[10];
  const float* wkp   = (const float*)d_in[11];

  float* out_pred = (float*)d_out;
  float* out_A    = out_pred + (size_t)BT*NV;   // tuple: (pred, A) concatenated

  hipLaunchKernelGGL(psm_fused_kernel, dim3(BLOCKS), dim3(THREADS), 0, stream,
                     x, hist, ce_w1, ce_b1, ce_w2, ce_b2,
                     p_w1, p_b1, p_w2, p_b2, wqp, wkp, out_pred, out_A);
}
